// BahdanauAttention_10977936408726
// MI455X (gfx1250) — matrix operand order repaired and checked by reading, not compiled, not run on hardware
//
#include <hip/hip_runtime.h>
#include <hip/hip_bf16.h>

// ---------------------------------------------------------------------------
// Bahdanau attention, fused for MI455X (gfx1250, wave32, WMMA + async LDS).
// B=32, S=2048, H=K=1024. HBM floor ~512MB (~22us @ 23.3TB/s).
// ---------------------------------------------------------------------------

#define B_DIM 32
#define S_DIM 2048
#define H_DIM 1024
#define K_DIM 1024

typedef __attribute__((ext_vector_type(16))) __bf16 v16bf;
typedef __attribute__((ext_vector_type(8)))  float  v8f;

__device__ __forceinline__ __bf16 f2bf(float x) { return (__bf16)x; }

// Fast tanh via hardware v_exp_f32 + v_rcp_f32 (~5 VALU ops vs ~20+ for ocml).
// Saturates correctly: exp(2x)->inf -> rcp->0 -> tanh->1 (and -1 via 0).
__device__ __forceinline__ float fast_tanh(float x) {
    float e = __expf(2.0f * x);
    return 1.0f - 2.0f * __builtin_amdgcn_rcpf(e + 1.0f);
}

// ---------------------------------------------------------------------------
// Kernel 0: one-time Wc transpose+convert: WcT[h*K + n] = bf16(Wc_w[n*H + h]).
// 4MB read + 2MB write, negligible vs 512MB streaming floor.
// ---------------------------------------------------------------------------
__global__ void convert_wc_kernel(const float* __restrict__ Wc_w,
                                  __bf16* __restrict__ WcT) {
    int o = blockIdx.x * blockDim.x + threadIdx.x;   // 1M elements
    int h = o >> 10;
    int n = o & (K_DIM - 1);
    WcT[o] = f2bf(Wc_w[(size_t)n * H_DIM + h]);
}

// ---------------------------------------------------------------------------
// Kernel 1: dec_proj[b,k] = decoder_state[b,:] . Wb_w[k,:] + Wb_b[k]
// ---------------------------------------------------------------------------
__global__ void dec_proj_kernel(const float* __restrict__ dec,
                                const float* __restrict__ Wb_w,
                                const float* __restrict__ Wb_b,
                                float* __restrict__ dec_proj) {
    int idx = blockIdx.x * blockDim.x + threadIdx.x;   // 32768 threads
    int b = idx >> 10;
    int k = idx & (K_DIM - 1);
    const float4* dv = (const float4*)(dec + (size_t)b * H_DIM);
    const float4* wv = (const float4*)(Wb_w + (size_t)k * H_DIM);
    float s = Wb_b[k];
    #pragma unroll 8
    for (int i = 0; i < H_DIM / 4; ++i) {
        float4 d = dv[i];
        float4 w = wv[i];
        s = fmaf(d.x, w.x, s);
        s = fmaf(d.y, w.y, s);
        s = fmaf(d.z, w.z, s);
        s = fmaf(d.w, w.w, s);
    }
    dec_proj[idx] = s;
}

// ---------------------------------------------------------------------------
// Kernel 2 (WMMA + async-to-LDS):
//   scores[b,s] = sum_k Wa[k]*tanh(dec_proj[b,k]+Wc_b[k]+sum_h enc[b,s,h]Wc[k,h])
// One workgroup = (batch b, 32-row S tile). 8 waves; wave w owns features
// [w*128, w*128+128) for BOTH 16-row M-subtiles -> 16 v8f accumulator tiles
// (128 VGPRs). H loop in K=32 chunks; the WcT bf16 chunk (32x1024 = 64KB,
// fully contiguous) is staged into LDS with GLOBAL_LOAD_ASYNC_TO_LDS_B128.
// Each B fragment feeds two WMMAs (both M-subtiles).
// ---------------------------------------------------------------------------
__global__ void
__launch_bounds__(256)
score_kernel(const float* __restrict__ enc,       // [B,S,H]
             const __bf16* __restrict__ WcT,      // [H,K] bf16
             const float* __restrict__ Wc_b,      // [K]
             const float* __restrict__ dec_proj,  // [B,K]
             const float* __restrict__ Wa_w,      // [K]
             const float* __restrict__ Wa_b,      // [1]
             float* __restrict__ scores) {        // [B,S]
    __shared__ unsigned char smem[32 * K_DIM * 2] __attribute__((aligned(16)));
    __bf16* ldsB = (__bf16*)smem;                 // [32 h'][1024 n]

    const int tid  = threadIdx.x;
    const int lane = tid & 31;
    const int wave = tid >> 5;
    const int half = lane >> 4;          // 0 or 1
    const int m    = lane & 15;

    const int b  = blockIdx.x >> 6;              // S_DIM/32 = 64 tiles/batch
    const int s0 = (blockIdx.x & 63) << 5;       // 32-row M tile

    const int nbase_wave = wave * 128;

    const v8f zero8 = {0.f, 0.f, 0.f, 0.f, 0.f, 0.f, 0.f, 0.f};
    v8f acc0[8], acc1[8];
    #pragma unroll
    for (int t = 0; t < 8; ++t) { acc0[t] = zero8; acc1[t] = zero8; }

    // Per-lane A row pointers (16-bit A layout: lanes 0-15 use K={0..7,16..23},
    // lanes 16-31 use K={8..15,24..31} of each 32-wide chunk).
    const float* arow0 = enc + ((size_t)b * S_DIM + (s0 + m)) * H_DIM;
    const float* arow1 = arow0 + 16 * H_DIM;

    // LDS byte address (low 32 bits of the generic pointer == LDS offset).
    const unsigned lds_base = (unsigned)(unsigned long long)(void*)smem;
    const unsigned lane_off = (unsigned)tid * 16u;   // 16B per lane per issue

    for (int h0 = 0; h0 < H_DIM; h0 += 32) {
        // ---- async-copy WcT[h0:h0+32][:] (contiguous 64KB) into LDS ----
        {
            unsigned long long gsrc =
                (unsigned long long)(const void*)(WcT + (size_t)h0 * K_DIM);
            #pragma unroll
            for (int it = 0; it < 16; ++it) {
                unsigned off = lane_off + (unsigned)it * 4096u;  // 256t*16B
                asm volatile(
                    "global_load_async_to_lds_b128 %0, %1, %2"
                    :
                    : "v"(lds_base + off), "v"(off), "s"(gsrc)
                    : "memory");
            }
            asm volatile("s_wait_asynccnt 0x0" ::: "memory");
        }
        __syncthreads();

        if (h0 + 32 < H_DIM)
            __builtin_prefetch(arow0 + h0 + 32, 0, 0);

        // ---- build two A fragments from global (contiguous 8-float runs) ----
        v16bf aF0, aF1;
        #pragma unroll
        for (int j = 0; j < 8; ++j) {
            aF0[j]     = f2bf(arow0[h0 + half * 8 + j]);
            aF0[8 + j] = f2bf(arow0[h0 + 16 + half * 8 + j]);
            aF1[j]     = f2bf(arow1[h0 + half * 8 + j]);
            aF1[8 + j] = f2bf(arow1[h0 + 16 + half * 8 + j]);
        }

        // ---- 16 WMMAs: each B fragment reused for both M-subtiles ----
        const int k_b = 8 * half;    // B-fragment K base (table order, same as A)
        #pragma unroll
        for (int t = 0; t < 8; ++t) {
            const __bf16* bp = ldsB + k_b * K_DIM + nbase_wave + t * 16 + m;
            v16bf bF;
            #pragma unroll
            for (int j = 0; j < 8; ++j) {
                bF[j]     = bp[j * K_DIM];
                bF[8 + j] = bp[(16 + j) * K_DIM];
            }
            acc0[t] = __builtin_amdgcn_wmma_f32_16x16x32_bf16(
                false, aF0, false, bF, (short)0, acc0[t], false, false);
            acc1[t] = __builtin_amdgcn_wmma_f32_16x16x32_bf16(
                false, aF1, false, bF, (short)0, acc1[t], false, false);
        }
        __syncthreads();   // ldsB consumed; safe to restage next chunk
    }

    // ---- epilogue: tanh, Wa dot, per-row partial sums ----
    // C layout: lane holds col N = nbase + t*16 + m; row M = v + 8*half
    // (subtile 0) and 16 + v + 8*half (subtile 1).
    float p0[8], p1[8];
    #pragma unroll
    for (int v = 0; v < 8; ++v) { p0[v] = 0.f; p1[v] = 0.f; }

    #pragma unroll
    for (int t = 0; t < 8; ++t) {
        const int col = nbase_wave + t * 16 + m;
        const float bias = dec_proj[(size_t)b * K_DIM + col] + Wc_b[col];
        const float wa = Wa_w[col];
        #pragma unroll
        for (int v = 0; v < 8; ++v) {
            p0[v] = fmaf(fast_tanh(acc0[t][v] + bias), wa, p0[v]);
            p1[v] = fmaf(fast_tanh(acc1[t][v] + bias), wa, p1[v]);
        }
    }

    // ---- reduce 32 row sums across all lanes via LDS float atomics ----
    float* lds_score = (float*)smem;   // reuse staging LDS
    if (tid < 32) lds_score[tid] = 0.f;
    __syncthreads();
    #pragma unroll
    for (int v = 0; v < 8; ++v) {
        atomicAdd(&lds_score[v + 8 * half], p0[v]);
        atomicAdd(&lds_score[16 + v + 8 * half], p1[v]);
    }
    __syncthreads();
    if (tid < 32)
        scores[(size_t)b * S_DIM + s0 + tid] = lds_score[tid] + Wa_b[0];
}

// ---------------------------------------------------------------------------
// Kernel 3: softmax over S per batch row. One block per batch.
// exp computed once per element (stored to attn, then normalized in place).
// ---------------------------------------------------------------------------
__global__ void softmax_kernel(const float* __restrict__ scores,
                               float* __restrict__ attn) {
    __shared__ float red[256];
    const int b = blockIdx.x;
    const int tid = threadIdx.x;
    const float* row = scores + (size_t)b * S_DIM;
    float* orow = attn + (size_t)b * S_DIM;

    float mx = -3.4e38f;
    for (int i = tid; i < S_DIM; i += 256) mx = fmaxf(mx, row[i]);
    red[tid] = mx;
    __syncthreads();
    for (int off = 128; off > 0; off >>= 1) {
        if (tid < off) red[tid] = fmaxf(red[tid], red[tid + off]);
        __syncthreads();
    }
    mx = red[0];
    __syncthreads();

    float sum = 0.f;
    for (int i = tid; i < S_DIM; i += 256) {
        float e = __expf(row[i] - mx);
        orow[i] = e;
        sum += e;
    }
    red[tid] = sum;
    __syncthreads();
    for (int off = 128; off > 0; off >>= 1) {
        if (tid < off) red[tid] += red[tid + off];
        __syncthreads();
    }
    const float inv = 1.0f / red[0];

    for (int i = tid; i < S_DIM; i += 256)
        orow[i] *= inv;
}

// ---------------------------------------------------------------------------
// Kernel 4a: zero output (harness poisons d_out).
// ---------------------------------------------------------------------------
__global__ void zero_kernel(float* __restrict__ out) {
    int idx = blockIdx.x * blockDim.x + threadIdx.x;
    if (idx < B_DIM * H_DIM) out[idx] = 0.f;
}

// ---------------------------------------------------------------------------
// Kernel 4b: context[b,h] = sum_s attn[b,s] * enc[b,s,h].
// Grid = B * 8 S-chunks; coalesced streaming; partials via global atomics.
// ---------------------------------------------------------------------------
__global__ void
__launch_bounds__(256)
context_kernel(const float* __restrict__ attn,
               const float* __restrict__ enc,
               float* __restrict__ out) {
    const int b  = blockIdx.x >> 3;
    const int sc = blockIdx.x & 7;
    const int tid = threadIdx.x;
    const int s_begin = sc * (S_DIM / 8);

    float acc0 = 0.f, acc1 = 0.f, acc2 = 0.f, acc3 = 0.f;
    for (int s = s_begin; s < s_begin + S_DIM / 8; ++s) {
        const float a = attn[(size_t)b * S_DIM + s];
        const float* row = enc + ((size_t)b * S_DIM + s) * H_DIM;
        acc0 = fmaf(a, row[tid      ], acc0);
        acc1 = fmaf(a, row[tid + 256], acc1);
        acc2 = fmaf(a, row[tid + 512], acc2);
        acc3 = fmaf(a, row[tid + 768], acc3);
    }
    float* orow = out + (size_t)b * H_DIM;
    atomicAdd(&orow[tid      ], acc0);
    atomicAdd(&orow[tid + 256], acc1);
    atomicAdd(&orow[tid + 512], acc2);
    atomicAdd(&orow[tid + 768], acc3);
}

// ---------------------------------------------------------------------------
// Launch
// ---------------------------------------------------------------------------
extern "C" void kernel_launch(void* const* d_in, const int* in_sizes, int n_in,
                              void* d_out, int out_size, void* d_ws, size_t ws_size,
                              hipStream_t stream) {
    const float* dec   = (const float*)d_in[0];  // [32,1024]
    const float* enc   = (const float*)d_in[1];  // [32,2048,1024]
    const float* Wa_w  = (const float*)d_in[2];  // [1,1024]
    const float* Wa_b  = (const float*)d_in[3];  // [1]
    const float* Wb_w  = (const float*)d_in[4];  // [1024,1024]
    const float* Wb_b  = (const float*)d_in[5];  // [1024]
    const float* Wc_w  = (const float*)d_in[6];  // [1024,1024]
    const float* Wc_b  = (const float*)d_in[7];  // [1024]
    float* out = (float*)d_out;                  // [32,1024]

    // workspace layout: WcT (2MB bf16) | dec_proj | scores | attn
    __bf16* WcT      = (__bf16*)d_ws;
    float* dec_proj  = (float*)((char*)d_ws + (size_t)H_DIM * K_DIM * 2);
    float* scores    = dec_proj + B_DIM * K_DIM;
    float* attn      = scores + B_DIM * S_DIM;

    convert_wc_kernel<<<(H_DIM * K_DIM) / 256, 256, 0, stream>>>(Wc_w, WcT);
    dec_proj_kernel<<<(B_DIM * K_DIM) / 256, 256, 0, stream>>>(dec, Wb_w, Wb_b, dec_proj);
    score_kernel<<<B_DIM * (S_DIM / 32), 256, 0, stream>>>(enc, WcT, Wc_b, dec_proj,
                                                           Wa_w, Wa_b, scores);
    softmax_kernel<<<B_DIM, 256, 0, stream>>>(scores, attn);
    zero_kernel<<<(B_DIM * H_DIM) / 256, 256, 0, stream>>>(out);
    context_kernel<<<B_DIM * 8, 256, 0, stream>>>(attn, enc, out);
}
